// CgpHmmCell_27865747816972
// MI455X (gfx1250) — matrix operand order repaired and checked by reading, compiled, not run on hardware
//
#include <hip/hip_runtime.h>
#include <hip/hip_bf16.h>

// ---------------- problem constants (match reference) ----------------
#define S_STATES 612
#define N_COLS   100
#define AB       4
#define NG       (N_COLS / AB)   // 25
#define NWAVES   20              // 640 threads / 32
#define BATCH    32

typedef __attribute__((ext_vector_type(16))) _Float16 v16h;
typedef __attribute__((ext_vector_type(8)))  float    v8f;

// =====================================================================
// Kernel 1: build softmaxed sparse parameters in workspace.
//   Bm   : [COLS][S] row-major (emission row for an observed column is contiguous)
//   Aval : [S][4]   row-softmax of transition kernel (offsets {0,1,5,17})
//   Iv   : [4]      softmax of init kernel
// =====================================================================
__global__ void cgphmm_prep_kernel(const float* __restrict__ transk,
                                   const float* __restrict__ emisk,
                                   const float* __restrict__ initk,
                                   float* __restrict__ Bm,
                                   float* __restrict__ Aval,
                                   float* __restrict__ Iv) {
    const int tid = threadIdx.x;
    const int nthr = blockDim.x;

    // zero the dense emission matrix (scatter touches only 12/100 cols per state)
    for (int i = tid; i < N_COLS * S_STATES; i += nthr) Bm[i] = 0.0f;

    // transition row softmax: consecutive groups of 4 in transk
    for (int g = tid; g < S_STATES; g += nthr) {
        float v0 = transk[4 * g + 0], v1 = transk[4 * g + 1];
        float v2 = transk[4 * g + 2], v3 = transk[4 * g + 3];
        float m = fmaxf(fmaxf(v0, v1), fmaxf(v2, v3));
        float e0 = __expf(v0 - m), e1 = __expf(v1 - m);
        float e2 = __expf(v2 - m), e3 = __expf(v3 - m);
        float inv = 1.0f / (e0 + e1 + e2 + e3);
        Aval[4 * g + 0] = e0 * inv;
        Aval[4 * g + 1] = e1 * inv;
        Aval[4 * g + 2] = e2 * inv;
        Aval[4 * g + 3] = e3 * inv;
    }
    __syncthreads();   // Bm zeroing must complete before the scatter below

    // emission softmax: consecutive groups of 4 in emisk form one (state,group) segment
    for (int g = tid; g < S_STATES * 3; g += nthr) {
        int s = g / 3;
        int o = g - 3 * s;                    // 0,1,2
        int offG = (o == 0) ? 0 : (o == 1) ? 7 : 13;
        int grp  = (s + offG) % NG;
        int c0   = grp * AB;                  // base column
        float v0 = emisk[4 * g + 0], v1 = emisk[4 * g + 1];
        float v2 = emisk[4 * g + 2], v3 = emisk[4 * g + 3];
        float m = fmaxf(fmaxf(v0, v1), fmaxf(v2, v3));
        float e0 = __expf(v0 - m), e1 = __expf(v1 - m);
        float e2 = __expf(v2 - m), e3 = __expf(v3 - m);
        float inv = 1.0f / (e0 + e1 + e2 + e3);
        Bm[(size_t)(c0 + 0) * S_STATES + s] = e0 * inv;
        Bm[(size_t)(c0 + 1) * S_STATES + s] = e1 * inv;
        Bm[(size_t)(c0 + 2) * S_STATES + s] = e2 * inv;
        Bm[(size_t)(c0 + 3) * S_STATES + s] = e3 * inv;
    }

    if (tid == 0) {
        float v0 = initk[0], v1 = initk[1], v2 = initk[2], v3 = initk[3];
        float m = fmaxf(fmaxf(v0, v1), fmaxf(v2, v3));
        float e0 = __expf(v0 - m), e1 = __expf(v1 - m);
        float e2 = __expf(v2 - m), e3 = __expf(v3 - m);
        float inv = 1.0f / (e0 + e1 + e2 + e3);
        Iv[0] = e0 * inv; Iv[1] = e1 * inv; Iv[2] = e2 * inv; Iv[3] = e3 * inv;
    }
}

// =====================================================================
// Kernel 2: one-hot -> observation index via WMMA (streaming pass, 52 MB).
//   obs_row = onehot_row(100) . iota(100)  — exact in f16*f16+f32.
//   One wave handles a 16-row tile with 4 chained v_wmma_f32_16x16x32_f16.
//   A-tile (16x32 f16): lane l (hi = l>=16, m = l&15) holds row m,
//     K = {k0..k0+7, k0+16..k0+23}, k0 = 8*hi  (two aligned float4 pairs).
//   B-tile (32x16 f16): n = lane&15, element i -> K = (i&7) + 8*hi + 16*(i>>3) (+32 per chunk).
//     Column 0 = iota, other columns zero; built ONCE, advanced by +32/chunk
//     (exact f16 integer adds). No masking needed past col 99: the A tile is
//     zero there, so those products vanish.
//   D (16x16 f32): lane 0 holds (M=0..7, N=0) in acc[0..7], lane 16 holds M=8..15.
// =====================================================================
__global__ void cgphmm_obs_wmma_kernel(const float* __restrict__ onehot,
                                       int* __restrict__ obs,
                                       int nrows) {
    const int lane  = threadIdx.x & 31;
    const int gwave = (int)((blockIdx.x * blockDim.x + threadIdx.x) >> 5);
    const int row0  = gwave * 16;
    if (row0 >= nrows) return;                 // wave-uniform: EXEC stays all-ones

    const int n  = lane & 15;
    const int hi = lane >> 4;                  // 0 or 1
    // row base is 400 B = 25 * 16 B  ->  every float4 below is 16-B aligned
    const float4* rowp4 = (const float4*)(onehot + (size_t)(row0 + n) * N_COLS);

    // ---- B tile (chunk 0) + per-chunk increment ----
    const _Float16 inc = (n == 0) ? (_Float16)32.0f : (_Float16)0.0f;
    v16h b;
#pragma unroll
    for (int i = 0; i < 16; ++i)
        b[i] = (n == 0) ? (_Float16)(float)((i & 7) + 8 * hi + 16 * (i >> 3)) : (_Float16)0.0f;

    v8f acc = {};
#pragma unroll
    for (int c = 0; c < 4; ++c) {
        v16h a;
        if (c < 3) {
            // cols [k0, k0+8) and [k0+16, k0+24), all < 96 < 100: unconditional
            const int q = (32 * c + 8 * hi) >> 2;     // float4 index of run 1
            const float4 f0 = rowp4[q + 0];
            const float4 f1 = rowp4[q + 1];
            const float4 f2 = rowp4[q + 4];
            const float4 f3 = rowp4[q + 5];
            a[0]  = (_Float16)f0.x; a[1]  = (_Float16)f0.y;
            a[2]  = (_Float16)f0.z; a[3]  = (_Float16)f0.w;
            a[4]  = (_Float16)f1.x; a[5]  = (_Float16)f1.y;
            a[6]  = (_Float16)f1.z; a[7]  = (_Float16)f1.w;
            a[8]  = (_Float16)f2.x; a[9]  = (_Float16)f2.y;
            a[10] = (_Float16)f2.z; a[11] = (_Float16)f2.w;
            a[12] = (_Float16)f3.x; a[13] = (_Float16)f3.y;
            a[14] = (_Float16)f3.z; a[15] = (_Float16)f3.w;
        } else {
            // chunk 3: only cols 96..99 exist; they live in the first float4 of
            // the hi==0 run. Everything else is zero.
#pragma unroll
            for (int i = 0; i < 16; ++i) a[i] = (_Float16)0.0f;
            if (hi == 0) {                     // EXEC-masked load, always in-bounds
                const float4 f0 = rowp4[24];   // cols 96..99
                a[0] = (_Float16)f0.x; a[1] = (_Float16)f0.y;
                a[2] = (_Float16)f0.z; a[3] = (_Float16)f0.w;
            }
        }
        acc = __builtin_amdgcn_wmma_f32_16x16x32_f16(
                  false, a, false, b, (short)0, acc, false, false);
        if (c < 3) {
#pragma unroll
            for (int i = 0; i < 16; ++i) b[i] = b[i] + inc;   // exact (<128)
        }
    }

    if (n == 0) {
#pragma unroll
        for (int r = 0; r < 8; ++r) {
            const int row = row0 + r + 8 * hi;
            if (row < nrows) obs[row] = (int)(acc[r] + 0.5f);
        }
    }
}

// =====================================================================
// Kernel 3: sequential forward scan. One workgroup per batch element,
// 640 threads (20 wave32s), one state per thread, alpha in LDS.
// =====================================================================
__global__ __launch_bounds__(640)
void cgphmm_forward_scan_kernel(const float* __restrict__ Bm,
                                const float* __restrict__ Aval,
                                const float* __restrict__ Iv,
                                const int* __restrict__ obs,
                                float* __restrict__ alphaOut,
                                float* __restrict__ llOut,
                                int T) {
    __shared__ float aLds[S_STATES];
    __shared__ float wLds[NWAVES];
    __shared__ float zLds;

    const int b    = blockIdx.x;
    const int j    = threadIdx.x;
    const int wave = j >> 5;
    const int lane = j & 31;
    const bool act = (j < S_STATES);
    const int* obsb = obs + (size_t)b * T;

    // incoming-edge indices: column j receives from i_k = (j - off_k) mod S
    int i0 = j;
    int i1 = j - 1;  if (i1 < 0) i1 += S_STATES;
    int i2 = j - 5;  if (i2 < 0) i2 += S_STATES;
    int i3 = j - 17; if (i3 < 0) i3 += S_STATES;
    float w0 = 0.f, w1 = 0.f, w2 = 0.f, w3 = 0.f;
    if (act) {
        w0 = Aval[i0 * 4 + 0];
        w1 = Aval[i1 * 4 + 1];
        w2 = Aval[i2 * 4 + 2];
        w3 = Aval[i3 * 4 + 3];
    }

    // t = 0 : alpha = Bm[obs0,:] * I
    float a = 0.0f;
    if (act) {
        const float iv = (j < 4) ? Iv[j] : 0.0f;
        a = Bm[(size_t)obsb[0] * S_STATES + j] * iv;
    }
    float ll = 0.0f;

    for (int t = 1; t < T; ++t) {
        if (act) aLds[j] = a;
        // wave-level partial sum of alpha (inactive threads contribute 0)
        float p = a;
#pragma unroll
        for (int s = 16; s > 0; s >>= 1) p += __shfl_down(p, s, 32);
        if (lane == 0) wLds[wave] = p;
        __syncthreads();

        if (j == 0) {
            float Z = 0.0f;
#pragma unroll
            for (int w = 0; w < NWAVES; ++w) Z += wLds[w];
            zLds = Z;
        }
        __syncthreads();

        const float Z = zLds;
        if (j == 0) ll += __logf(Z);

        if (act) {
            const float Zinv = 1.0f / Z;
            const float R = aLds[i0] * w0 + aLds[i1] * w1
                          + aLds[i2] * w2 + aLds[i3] * w3;
            const float* brow = Bm + (size_t)obsb[t] * S_STATES;
            a = brow[j] * R * Zinv;
            if (t + 1 < T) {
                // warm L2/L0 for the next emission row (global_prefetch_b8)
                __builtin_prefetch(Bm + (size_t)obsb[t + 1] * S_STATES + j, 0, 1);
            }
        }
        __syncthreads();   // protect aLds reads before next iteration's stores
    }

    // final outputs
    if (act) alphaOut[(size_t)b * S_STATES + j] = a;
    float p = a;
#pragma unroll
    for (int s = 16; s > 0; s >>= 1) p += __shfl_down(p, s, 32);
    if (lane == 0) wLds[wave] = p;
    __syncthreads();
    if (j == 0) {
        float Z = 0.0f;
#pragma unroll
        for (int w = 0; w < NWAVES; ++w) Z += wLds[w];
        llOut[b] = ll + __logf(Z);
    }
}

// =====================================================================
// Host-side launcher
// Inputs (setup_inputs order):
//   0 inputs [B,T,COLS] f32, 1 init_kernel[4] f32, 2 transition_kernel[S*4] f32,
//   3 emission_kernel[S*12] f32, 4..8 index arrays (unused, pattern is static)
// Output: alphaT [B,S] f32 then loglik [B] f32, concatenated.
// =====================================================================
extern "C" void kernel_launch(void* const* d_in, const int* in_sizes, int n_in,
                              void* d_out, int out_size, void* d_ws, size_t ws_size,
                              hipStream_t stream) {
    const float* inputs = (const float*)d_in[0];
    const float* initk  = (const float*)d_in[1];
    const float* transk = (const float*)d_in[2];
    const float* emisk  = (const float*)d_in[3];

    const int BT = in_sizes[0] / N_COLS;   // B*T = 131072
    const int B  = BATCH;
    const int T  = BT / B;                 // 4096

    // workspace layout (floats, then ints)
    float* ws    = (float*)d_ws;
    float* Bm    = ws;                                   // COLS*S
    float* Aval  = Bm + (size_t)N_COLS * S_STATES;       // S*4
    float* Iv    = Aval + (size_t)S_STATES * 4;          // 4 (padded to 8)
    int*   obsbf = (int*)(Iv + 8);                       // BT ints

    float* alphaOut = (float*)d_out;                     // B*S
    float* llOut    = alphaOut + (size_t)B * S_STATES;   // B

    // 1) softmaxes + dense scatter of emission matrix
    cgphmm_prep_kernel<<<1, 256, 0, stream>>>(transk, emisk, initk, Bm, Aval, Iv);

    // 2) one-hot -> obs index via WMMA (one wave per 16 rows)
    const int waves   = BT / 16;                 // 8192 (BT divisible by 16)
    const int threads = waves * 32;
    const int blocks  = (threads + 255) / 256;   // 1024
    cgphmm_obs_wmma_kernel<<<blocks, 256, 0, stream>>>(inputs, obsbf, BT);

    // 3) sequential forward scan, one workgroup per batch element
    cgphmm_forward_scan_kernel<<<B, 640, 0, stream>>>(Bm, Aval, Iv, obsbf,
                                                      alphaOut, llOut, T);
}
